// GCNNet_8993661518248
// MI455X (gfx1250) — matrix operand order repaired and checked by reading, not compiled, not run on hardware
//
#include <hip/hip_runtime.h>
#include <hip/hip_bf16.h>
#include <math.h>

#define HDIM 128
#define GNUM 512
#define CNUM 10

typedef __attribute__((ext_vector_type(16))) _Float16 v16h;
typedef __attribute__((ext_vector_type(8)))  float    v8f;

union AFrag { v16h v; _Float16 e[16]; };
union Acc8  { v8f  v; float    e[8];  };

__device__ __forceinline__ int imin(int a, int b) { return a < b ? a : b; }

// ---------------------------------------------------------------------------
// Elementwise / utility kernels
// ---------------------------------------------------------------------------
__global__ void zero_kernel(float* __restrict__ p, int n) {
    int i = blockIdx.x * blockDim.x + threadIdx.x;
    if (i < n) p[i] = 0.0f;
}

__global__ void fill1_kernel(float* __restrict__ p, int n) {
    int i = blockIdx.x * blockDim.x + threadIdx.x;
    if (i < n) p[i] = 1.0f;
}

__global__ void deg_accum_kernel(const int* __restrict__ dst, int E, float* __restrict__ deg) {
    int i = blockIdx.x * blockDim.x + threadIdx.x;
    if (i < E) atomicAdd(&deg[dst[i]], 1.0f);
}

__global__ void rsqrt_kernel(float* __restrict__ p, int n) {
    int i = blockIdx.x * blockDim.x + threadIdx.x;
    if (i < n) p[i] = rsqrtf(p[i]);
}

// ---------------------------------------------------------------------------
// BatchNorm: column sums (atomic partials) -> finalize into scale/shift
// ---------------------------------------------------------------------------
__global__ void bn_accum_kernel(const float* __restrict__ x, int rows,
                                float* __restrict__ sum, float* __restrict__ sq) {
    int f   = threadIdx.x & (HDIM - 1);
    int sub = threadIdx.x >> 7;            // 0 or 1
    int r0   = blockIdx.x * 1024 + sub;
    int rend = imin(blockIdx.x * 1024 + 1024, rows);
    float s = 0.0f, q = 0.0f;
    for (int r = r0; r < rend; r += 2) {
        float v = x[(size_t)r * HDIM + f];
        s += v; q += v * v;
    }
    atomicAdd(&sum[f], s);
    atomicAdd(&sq[f],  q);
}

__global__ void bn_finalize_kernel(const float* __restrict__ sum, const float* __restrict__ sq,
                                   float rowsInv,
                                   const float* __restrict__ g, const float* __restrict__ b,
                                   float* __restrict__ scale, float* __restrict__ shift) {
    int f = threadIdx.x;                   // 128 threads
    float mu   = sum[f] * rowsInv;
    float var  = sq[f]  * rowsInv - mu * mu;
    float rstd = rsqrtf(var + 1e-5f);
    float sc   = g[f] * rstd;
    scale[f] = sc;
    shift[f] = b[f] - mu * sc;
}

// Convert f32 weights W[k][n] (128x128 row-major) to transposed f16 WT[n][k]
__global__ void cvt_transpose_kernel(const float* __restrict__ W, _Float16* __restrict__ WT) {
    int i = blockIdx.x * blockDim.x + threadIdx.x;   // 16384 threads
    if (i < HDIM * HDIM) {
        int k = i >> 7, n = i & (HDIM - 1);
        WT[n * HDIM + k] = (_Float16)W[i];
    }
}

// ---------------------------------------------------------------------------
// Fused BN + WMMA GEMM:
//   C[M,128] = BN(A_f32[M,128]) @ W[128,128]    (optionally + bias, ReLU)
// BN is applied in-register on the A fragment: a = A*scale[k] + shift[k],
// converted to f16. Weights come pre-transposed (WT[n][k]) and are staged
// into LDS with CDNA5 async-to-LDS b128 copies (ASYNCcnt tracked).
// 256 threads = 8 waves; wave w handles rows [blk*128 + w*16, +16).
// ---------------------------------------------------------------------------
__global__ __launch_bounds__(256) void gemm128_bn_wmma_kernel(
        const float* __restrict__ A,
        const float* __restrict__ scale, const float* __restrict__ shift,
        const _Float16* __restrict__ WT,
        float* __restrict__ C, int M, const float* __restrict__ bias, int relu) {
    // Single shared object -> sits at LDS offset 0; async VDST operand is the
    // plain byte offset into this struct. 32-byte alignment lets the B
    // fragment reads lower to ds_load_b128 pairs.
    __shared__ struct alignas(32) {
        _Float16 w[HDIM * HDIM];   // 32 KB, WT layout [n][k], at offset 0
        float    sc[HDIM];
        float    sh[HDIM];
    } smem;

    int tid = threadIdx.x;

    // Async stage of WT (32 KB) into LDS: 2048 x b128, 8 chunks per thread.
    for (int i = tid; i < (HDIM * HDIM * 2) / 16; i += 256) {
        unsigned ldsOff = (unsigned)(i * 16);          // byte offset of smem.w[i*8]
        const _Float16* gp = WT + (size_t)i * 8;
        asm volatile("global_load_async_to_lds_b128 %0, %1, off"
                     :: "v"(ldsOff), "v"(gp) : "memory");
    }
    // Stage BN scale/shift (1 KB) with plain loads.
    if (tid < HDIM) smem.sc[tid] = scale[tid];
    else            smem.sh[tid - HDIM] = shift[tid - HDIM];
    asm volatile("s_wait_asynccnt 0x0" ::: "memory");  // this wave's asyncs done
    __syncthreads();                                   // all waves' staging visible

    int lane = tid & 31;
    int wave = tid >> 5;
    int r    = lane & 15;      // A row / D column selector
    int hf   = lane >> 4;      // half selector (K sub-range)
    int rowBase = blockIdx.x * 128 + wave * 16;

    Acc8 acc[8];
    #pragma unroll
    for (int nt = 0; nt < 8; ++nt)
        #pragma unroll
        for (int i = 0; i < 8; ++i) acc[nt].e[i] = 0.0f;

    #pragma unroll
    for (int ks = 0; ks < 4; ++ks) {
        int kbase = ks * 32;
        // A fragment (16x32 f16): lane holds row M=r, K runs {k0..k0+15}
        AFrag af;
        int arow = rowBase + r;
        int k0   = kbase + hf * 16;
        if (arow < M) {
            const float* ap = (const float*)__builtin_assume_aligned(
                A + (size_t)arow * HDIM + k0, 32);
            #pragma unroll
            for (int i = 0; i < 8; ++i) {
                af.e[i]     = (_Float16)(ap[i]      * smem.sc[k0 + i]      + smem.sh[k0 + i]);
                af.e[8 + i] = (_Float16)(ap[8 + i]  * smem.sc[k0 + 8 + i]  + smem.sh[k0 + 8 + i]);
            }
        } else {
            #pragma unroll
            for (int i = 0; i < 16; ++i) af.e[i] = (_Float16)0.0f;
        }
        // 8 output column-tiles
        #pragma unroll
        for (int nt = 0; nt < 8; ++nt) {
            // B fragment (32x16): lane holds col N=r, K = kbase + hf*16 + i.
            // Address = (nt*16+r)*256 + (kbase+hf*16)*2 bytes -> 32B aligned.
            const v16h* bp = (const v16h*)__builtin_assume_aligned(
                &smem.w[(nt * 16 + r) * HDIM + kbase + hf * 16], 32);
            v16h bfrag = *bp;
            acc[nt].v = __builtin_amdgcn_wmma_f32_16x16x32_f16(
                false, af.v, false, bfrag, (short)0, acc[nt].v, false, false);
        }
    }

    // Epilogue: D layout — VGPR v, lanes 0-15 row M=v, lanes 16-31 row M=8+v; col = lane%16
    #pragma unroll
    for (int nt = 0; nt < 8; ++nt) {
        int col = nt * 16 + r;
        float bv = bias ? bias[col] : 0.0f;
        #pragma unroll
        for (int v = 0; v < 8; ++v) {
            int row = rowBase + hf * 8 + v;
            if (row < M) {
                float val = acc[nt].e[v] + bv;
                if (relu) val = val > 0.0f ? val : 0.0f;
                C[(size_t)row * HDIM + col] = val;
            }
        }
    }
}

// ---------------------------------------------------------------------------
// Graph propagation
// ---------------------------------------------------------------------------
__global__ void selfloop_kernel(const float* __restrict__ m, const float* __restrict__ dinv,
                                float* __restrict__ acc, int n) {
    int i = blockIdx.x * blockDim.x + threadIdx.x;
    if (i < n) {
        float d = dinv[i >> 7];
        acc[i] = m[i] * d * d;
    }
}

__global__ void edge_scatter_kernel(const int* __restrict__ src, const int* __restrict__ dst, int E,
                                    const float* __restrict__ m, const float* __restrict__ dinv,
                                    float* __restrict__ acc) {
    long long t = (long long)blockIdx.x * blockDim.x + threadIdx.x;
    if (t >= (long long)E * 32) return;
    int e  = (int)(t >> 5);
    int f4 = (int)(t & 31);
    int s = src[e], d = dst[e];
    float w = dinv[s] * dinv[d];
    const float4 mv = *(const float4*)&m[(size_t)s * HDIM + f4 * 4];
    float* ap = &acc[(size_t)d * HDIM + f4 * 4];
    atomicAdd(ap + 0, mv.x * w);
    atomicAdd(ap + 1, mv.y * w);
    atomicAdd(ap + 2, mv.z * w);
    atomicAdd(ap + 3, mv.w * w);
}

__global__ void bias_relu_kernel(float* __restrict__ a, const float* __restrict__ bias, int n) {
    int i = blockIdx.x * blockDim.x + threadIdx.x;
    if (i < n) {
        float v = a[i] + bias[i & (HDIM - 1)];
        a[i] = v > 0.0f ? v : 0.0f;
    }
}

// ---------------------------------------------------------------------------
// Pooling
// ---------------------------------------------------------------------------
__global__ void pool_scatter_kernel(const float* __restrict__ h, const int* __restrict__ batch,
                                    float* __restrict__ pooled, int N) {
    long long t = (long long)blockIdx.x * blockDim.x + threadIdx.x;
    if (t >= (long long)N * 32) return;
    int i  = (int)(t >> 5);
    int f4 = (int)(t & 31);
    int g = batch[i];
    const float4 hv = *(const float4*)&h[(size_t)i * HDIM + f4 * 4];
    float* pp = &pooled[(size_t)g * HDIM + f4 * 4];
    atomicAdd(pp + 0, hv.x);
    atomicAdd(pp + 1, hv.y);
    atomicAdd(pp + 2, hv.z);
    atomicAdd(pp + 3, hv.w);
}

__global__ void count_kernel(const int* __restrict__ batch, float* __restrict__ counts, int N) {
    int i = blockIdx.x * blockDim.x + threadIdx.x;
    if (i < N) atomicAdd(&counts[batch[i]], 1.0f);
}

__global__ void pool_div_kernel(float* __restrict__ pooled, const float* __restrict__ counts, int n) {
    int i = blockIdx.x * blockDim.x + threadIdx.x;
    if (i < n) pooled[i] /= fmaxf(counts[i >> 7], 1.0f);
}

// ---------------------------------------------------------------------------
// Head: fused BN + final linear, then log-softmax
// ---------------------------------------------------------------------------
__global__ void logits_kernel(const float* __restrict__ hid, const float* __restrict__ scale,
                              const float* __restrict__ shift, const float* __restrict__ Wc,
                              const float* __restrict__ bc, float* __restrict__ logits, int G) {
    int t = blockIdx.x * blockDim.x + threadIdx.x;
    if (t >= G * CNUM) return;
    int g = t / CNUM, c = t % CNUM;
    float s = bc[c];
    for (int k = 0; k < HDIM; ++k)
        s += (hid[(size_t)g * HDIM + k] * scale[k] + shift[k]) * Wc[k * CNUM + c];
    logits[t] = s;
}

__global__ void logsoftmax_kernel(const float* __restrict__ logits, float* __restrict__ out, int G) {
    int g = blockIdx.x * blockDim.x + threadIdx.x;
    if (g >= G) return;
    float mx = -1e30f;
    for (int c = 0; c < CNUM; ++c) mx = fmaxf(mx, logits[g * CNUM + c]);
    float s = 0.0f;
    for (int c = 0; c < CNUM; ++c) s += expf(logits[g * CNUM + c] - mx);
    float lse = mx + logf(s);
    for (int c = 0; c < CNUM; ++c) out[g * CNUM + c] = logits[g * CNUM + c] - lse;
}

// ---------------------------------------------------------------------------
// Host launcher
// ---------------------------------------------------------------------------
extern "C" void kernel_launch(void* const* d_in, const int* in_sizes, int n_in,
                              void* d_out, int out_size, void* d_ws, size_t ws_size,
                              hipStream_t stream) {
    const int H = HDIM, G = GNUM, C = CNUM;
    const int N = in_sizes[0] / H;
    const int E = in_sizes[1] / 2;

    const float* x      = (const float*)d_in[0];
    const int*   ei     = (const int*)  d_in[1];
    const int*   batch  = (const int*)  d_in[2];
    const float* bnf_g  = (const float*)d_in[3];
    const float* bnf_b  = (const float*)d_in[4];
    const float* Wf     = (const float*)d_in[5];
    const float* bf     = (const float*)d_in[6];
    const float* convsW = (const float*)d_in[7];
    const float* convsB = (const float*)d_in[8];
    const float* bnc_g  = (const float*)d_in[9];
    const float* bnc_b  = (const float*)d_in[10];
    const float* bnfc_g = (const float*)d_in[11];
    const float* bnfc_b = (const float*)d_in[12];
    const float* linW   = (const float*)d_in[13];
    const float* linB   = (const float*)d_in[14];
    const float* bnh_g  = (const float*)d_in[15];
    const float* bnh_b  = (const float*)d_in[16];
    const float* Wc     = (const float*)d_in[17];
    const float* bc     = (const float*)d_in[18];

    const int* srcIdx = ei;
    const int* dstIdx = ei + E;

    // Workspace carve-out
    char* ws = (char*)d_ws;
    size_t off = 0;
    auto alloc = [&](size_t bytes) -> char* {
        char* p = ws + off;
        off += (bytes + 255) & ~(size_t)255;
        return p;
    };
    float*    h      = (float*)   alloc((size_t)N * H * 4);
    float*    m      = (float*)   alloc((size_t)N * H * 4);
    float*    dinv   = (float*)   alloc((size_t)N * 4);
    float*    stats  = (float*)   alloc((size_t)4 * H * 4);   // sum|sq|scale|shift
    _Float16* wT     = (_Float16*)alloc((size_t)H * H * 2);
    float*    pooled = (float*)   alloc((size_t)(G * H + G) * 4);
    float*    counts = pooled + (size_t)G * H;
    float*    hidden = (float*)   alloc((size_t)G * H * 4);
    float*    logits = (float*)   alloc((size_t)G * C * 4);
    float *sum = stats, *sq = stats + H, *scale = stats + 2 * H, *shift = stats + 3 * H;

    auto cdiv = [](long long a, long long b) -> int { return (int)((a + b - 1) / b); };
    const long long NH = (long long)N * H;

    // --- GCN normalization: dinv = deg^-1/2 (self-loop included) ---
    fill1_kernel<<<cdiv(N, 256), 256, 0, stream>>>(dinv, N);
    deg_accum_kernel<<<cdiv(E, 256), 256, 0, stream>>>(dstIdx, E, dinv);
    rsqrt_kernel<<<cdiv(N, 256), 256, 0, stream>>>(dinv, N);

    // --- BN(x) fused into feature linear (bias+ReLU) via WMMA ---
    zero_kernel<<<1, 256, 0, stream>>>(sum, 2 * H);
    bn_accum_kernel<<<cdiv(N, 1024), 256, 0, stream>>>(x, N, sum, sq);
    bn_finalize_kernel<<<1, H, 0, stream>>>(sum, sq, 1.0f / N, bnf_g, bnf_b, scale, shift);
    cvt_transpose_kernel<<<cdiv(H * H, 256), 256, 0, stream>>>(Wf, wT);
    gemm128_bn_wmma_kernel<<<cdiv(N, 128), 256, 0, stream>>>(x, scale, shift, wT, h, N, bf, 1);

    // --- 3 GCN conv layers ---
    for (int layer = 0; layer < 3; ++layer) {
        zero_kernel<<<1, 256, 0, stream>>>(sum, 2 * H);
        bn_accum_kernel<<<cdiv(N, 1024), 256, 0, stream>>>(h, N, sum, sq);
        bn_finalize_kernel<<<1, H, 0, stream>>>(sum, sq, 1.0f / N,
                                                bnc_g + layer * H, bnc_b + layer * H, scale, shift);
        cvt_transpose_kernel<<<cdiv(H * H, 256), 256, 0, stream>>>(convsW + (size_t)layer * H * H, wT);
        gemm128_bn_wmma_kernel<<<cdiv(N, 128), 256, 0, stream>>>(h, scale, shift, wT, m, N, nullptr, 0);
        // acc (reusing h) = self-loop term, then scatter edges, then bias+ReLU
        selfloop_kernel<<<cdiv(NH, 256), 256, 0, stream>>>(m, dinv, h, (int)NH);
        edge_scatter_kernel<<<cdiv((long long)E * 32, 256), 256, 0, stream>>>(srcIdx, dstIdx, E, m, dinv, h);
        bias_relu_kernel<<<cdiv(NH, 256), 256, 0, stream>>>(h, convsB + layer * H, (int)NH);
    }

    // --- Global mean pool ---
    zero_kernel<<<cdiv(G * H + G, 256), 256, 0, stream>>>(pooled, G * H + G);
    pool_scatter_kernel<<<cdiv((long long)N * 32, 256), 256, 0, stream>>>(h, batch, pooled, N);
    count_kernel<<<cdiv(N, 256), 256, 0, stream>>>(batch, counts, N);
    pool_div_kernel<<<cdiv(G * H, 256), 256, 0, stream>>>(pooled, counts, G * H);

    // --- Head: BN fused into linear(ReLU) via WMMA -> BN -> classifier -> log_softmax ---
    zero_kernel<<<1, 256, 0, stream>>>(sum, 2 * H);
    bn_accum_kernel<<<1, 256, 0, stream>>>(pooled, G, sum, sq);
    bn_finalize_kernel<<<1, H, 0, stream>>>(sum, sq, 1.0f / G, bnfc_g, bnfc_b, scale, shift);
    cvt_transpose_kernel<<<cdiv(H * H, 256), 256, 0, stream>>>(linW, wT);
    gemm128_bn_wmma_kernel<<<cdiv(G, 128), 256, 0, stream>>>(pooled, scale, shift, wT, hidden, G, linB, 1);

    zero_kernel<<<1, 256, 0, stream>>>(sum, 2 * H);
    bn_accum_kernel<<<1, 256, 0, stream>>>(hidden, G, sum, sq);
    bn_finalize_kernel<<<1, H, 0, stream>>>(sum, sq, 1.0f / G, bnh_g, bnh_b, scale, shift);
    logits_kernel<<<cdiv(G * C, 256), 256, 0, stream>>>(hidden, scale, shift, Wc, bc, logits, G);
    logsoftmax_kernel<<<cdiv(G, 256), 256, 0, stream>>>(logits, (float*)d_out, G);
}
